// MultiheadTemporalLatentAttention_77292231459360
// MI455X (gfx1250) — compile-verified, matrix operand order checked by reading, not run on hardware
//
#include <hip/hip_runtime.h>
#include <hip/hip_bf16.h>
#include <math.h>

// ---------------- problem constants ----------------
constexpr int HD   = 16;     // heads
constexpr int SD   = 2048;   // sequence
constexpr int BD   = 2;      // batch
constexpr int TT   = 1024;   // t = S/DR
constexpr int CCd  = 512;    // latent C
constexpr int DKC  = 192;    // NOPE+ROPE
constexpr int NOPEd= 128;
constexpr int ROPEd= 64;
constexpr int VHd  = 128;
constexpr int QRd  = 1536;
constexpr int EDIM = 2048;

typedef __attribute__((ext_vector_type(16))) __bf16        bf16x16;
typedef __attribute__((ext_vector_type(8)))  float         floatx8;
typedef __attribute__((ext_vector_type(8)))  unsigned int  uintx8;

__device__ __forceinline__ unsigned short f2bf(float f) {
  union { float f; unsigned int u; } v; v.f = f;
  unsigned int r = v.u + 0x7FFFu + ((v.u >> 16) & 1u);   // round-to-nearest-even
  return (unsigned short)(r >> 16);
}
__device__ __forceinline__ unsigned pk2(float lo, float hi) {
  return (unsigned)f2bf(lo) | ((unsigned)f2bf(hi) << 16);
}

// LDS byte address = low 32 bits of generic shared pointer (ISA 10.2).
__device__ __forceinline__ unsigned ldsoff(const void* p) {
  return (unsigned)(unsigned long long)p;
}
// CDNA5 async DMA: global -> LDS, 16B per active lane, tracked by ASYNCcnt.
__device__ __forceinline__ void async_b128(unsigned lds_byte, unsigned voff, const void* sbase) {
  asm volatile("global_load_async_to_lds_b128 %0, %1, %2"
               :: "v"(lds_byte), "v"(voff), "s"(sbase) : "memory");
}
__device__ __forceinline__ void wait_async0() {
  asm volatile("s_wait_asynccnt 0x0" ::: "memory");
}

// Fragment loader for 16-bit WMMA operands (A: lane=M row; B: lane=N col),
// per CDNA5 ISA 16-bit A layout: kb=(lane>>4)*8, halves [kb..kb+7],[kb+16..kb+23].
__device__ __forceinline__ bf16x16 load_frag(const unsigned short* p0, int pitch) {
  const int lane = threadIdx.x & 31;
  const unsigned short* p = p0 + (lane & 15) * pitch + ((lane >> 4) << 3);
  uint4 a = *reinterpret_cast<const uint4*>(p);
  uint4 b = *reinterpret_cast<const uint4*>(p + 16);
  uintx8 t;
  t[0]=a.x; t[1]=a.y; t[2]=a.z; t[3]=a.w;
  t[4]=b.x; t[5]=b.y; t[6]=b.z; t[7]=b.w;
  return __builtin_bit_cast(bf16x16, t);
}
__device__ __forceinline__ floatx8 wmma_bf16(bf16x16 a, bf16x16 b, floatx8 c) {
  return __builtin_amdgcn_wmma_f32_16x16x32_bf16(false, a, false, b, (short)0, c, false, false);
}

// ---------------- generic bf16 WMMA GEMM: Out[m,n] = sum_k A[m,k]*W[n,k] ----------------
// EPI: 0 plain f32, 1 +bias[n] f32, 2 bf16 split: n<128 -> Kcat (pitch 192),
//      n>=128 -> V transposed (b,h,d,t). Batched by blockIdx.z.
template <int EPI>
__global__ __launch_bounds__(256) void gemm_wmma_kernel(
    const float* __restrict__ A, const float* __restrict__ W,
    const float* __restrict__ bias, float* __restrict__ Out,
    int M, int N, int K, long sA, long sW, long sO,
    unsigned short* __restrict__ OutK, unsigned short* __restrict__ OutV)
{
  const int z = blockIdx.z;
  if (EPI == 2) {
    A += (long)(z / HD) * sA;     // A batched over b
    W += (long)(z % HD) * sW;     // weights batched over h
  } else {
    A += (long)z * sA; W += (long)z * sW; Out += (long)z * sO;
  }
  __shared__ __align__(16) unsigned short As[128][40];
  __shared__ __align__(16) unsigned short Bs[128][40];
  const int tid  = threadIdx.x;
  const int wave = tid >> 5;
  const int lane = tid & 31;
  const int waveM = wave & 3;   // 4 waves along M (32 rows each)
  const int waveN = wave >> 2;  // 2 waves along N (64 cols each)
  const int m0 = blockIdx.y * 128;
  const int n0 = blockIdx.x * 128;
  floatx8 acc[2][4] = {};

  const int lrow = tid >> 1;            // 0..127
  const int lcol = (tid & 1) * 16;      // 0 or 16

  for (int k0 = 0; k0 < K; k0 += 32) {
    {
      const int gm = m0 + lrow;
      const float* ap = A + (long)gm * K + k0 + lcol;
      float4 a4[4];
      if (gm < M) {
        const float4* p4 = reinterpret_cast<const float4*>(ap);
        a4[0]=p4[0]; a4[1]=p4[1]; a4[2]=p4[2]; a4[3]=p4[3];
      } else {
        a4[0]=a4[1]=a4[2]=a4[3]=make_float4(0.f,0.f,0.f,0.f);
      }
      uint4 lo, hi;
      lo.x=pk2(a4[0].x,a4[0].y); lo.y=pk2(a4[0].z,a4[0].w);
      lo.z=pk2(a4[1].x,a4[1].y); lo.w=pk2(a4[1].z,a4[1].w);
      hi.x=pk2(a4[2].x,a4[2].y); hi.y=pk2(a4[2].z,a4[2].w);
      hi.z=pk2(a4[3].x,a4[3].y); hi.w=pk2(a4[3].z,a4[3].w);
      *reinterpret_cast<uint4*>(&As[lrow][lcol])     = lo;
      *reinterpret_cast<uint4*>(&As[lrow][lcol + 8]) = hi;

      const int gn = n0 + lrow;
      const float* wp = W + (long)gn * K + k0 + lcol;
      if (gn < N) {
        const float4* p4 = reinterpret_cast<const float4*>(wp);
        a4[0]=p4[0]; a4[1]=p4[1]; a4[2]=p4[2]; a4[3]=p4[3];
      } else {
        a4[0]=a4[1]=a4[2]=a4[3]=make_float4(0.f,0.f,0.f,0.f);
      }
      lo.x=pk2(a4[0].x,a4[0].y); lo.y=pk2(a4[0].z,a4[0].w);
      lo.z=pk2(a4[1].x,a4[1].y); lo.w=pk2(a4[1].z,a4[1].w);
      hi.x=pk2(a4[2].x,a4[2].y); hi.y=pk2(a4[2].z,a4[2].w);
      hi.z=pk2(a4[3].x,a4[3].y); hi.w=pk2(a4[3].z,a4[3].w);
      *reinterpret_cast<uint4*>(&Bs[lrow][lcol])     = lo;
      *reinterpret_cast<uint4*>(&Bs[lrow][lcol + 8]) = hi;

      if (k0 + 32 < K) {                 // prefetch next tiles -> global_prefetch_b8
        if (gm < M) __builtin_prefetch(ap + 32, 0, 1);
        if (gn < N) __builtin_prefetch(wp + 32, 0, 1);
      }
    }
    __syncthreads();
    bf16x16 af[2], bfv[4];
    #pragma unroll
    for (int i = 0; i < 2; ++i) af[i]  = load_frag(&As[waveM * 32 + i * 16][0], 40);
    #pragma unroll
    for (int j = 0; j < 4; ++j) bfv[j] = load_frag(&Bs[waveN * 64 + j * 16][0], 40);
    #pragma unroll
    for (int i = 0; i < 2; ++i)
      #pragma unroll
      for (int j = 0; j < 4; ++j)
        acc[i][j] = wmma_bf16(af[i], bfv[j], acc[i][j]);
    __syncthreads();
  }

  const int cN = lane & 15;
  const int rB = (lane >> 4) << 3;
  #pragma unroll
  for (int i = 0; i < 2; ++i) {
    #pragma unroll
    for (int j = 0; j < 4; ++j) {
      #pragma unroll
      for (int e = 0; e < 8; ++e) {
        const int gm = m0 + waveM * 32 + i * 16 + rB + e;
        const int gn = n0 + waveN * 64 + j * 16 + cN;
        if (gm < M && gn < N) {
          float v = acc[i][j][e];
          if (EPI == 1) v += bias[gn];
          if (EPI == 2) {
            if (gn < NOPEd) OutK[((long)z * TT + gm) * DKC + gn] = f2bf(v);
            else            OutV[((long)z * VHd + (gn - NOPEd)) * TT + gm] = f2bf(v);
          } else {
            Out[(long)gm * N + gn] = v;
          }
        }
      }
    }
  }
}

// ---------------- LayerNorm over D (one block per row, in-place safe) ----------------
__global__ __launch_bounds__(256) void ln_kernel(
    const float* __restrict__ x, const float* __restrict__ g,
    const float* __restrict__ b, float* __restrict__ y, int D)
{
  const long row = blockIdx.x;
  const float* xr = x + row * D;
  float* yr = y + row * D;
  const int tid = threadIdx.x;
  __shared__ float red[256];
  float s = 0.f;
  for (int i = tid; i < D; i += 256) s += xr[i];
  red[tid] = s; __syncthreads();
  for (int off = 128; off > 0; off >>= 1) { if (tid < off) red[tid] += red[tid + off]; __syncthreads(); }
  const float mean = red[0] / (float)D; __syncthreads();
  float v = 0.f;
  for (int i = tid; i < D; i += 256) { float d = xr[i] - mean; v += d * d; }
  red[tid] = v; __syncthreads();
  for (int off = 128; off > 0; off >>= 1) { if (tid < off) red[tid] += red[tid + off]; __syncthreads(); }
  const float istd = rsqrtf(red[0] / (float)D + 1e-5f);
  for (int i = tid; i < D; i += 256) yr[i] = (xr[i] - mean) * istd * g[i] + b[i];
}

// ---------------- kv prep: LN(kvf[:, :512]) -> kvn ; rope(kvf[:, 512:576]) -> kpe -------
__global__ __launch_bounds__(256) void kvprep_kernel(
    const float* __restrict__ kvf, const float* __restrict__ g,
    const float* __restrict__ bb, float* __restrict__ kvn, float* __restrict__ kpe)
{
  const long r = blockIdx.x;                 // b*S + s
  const float* xr = kvf + r * 576;
  const int tid = threadIdx.x;
  if (tid < 32) {
    const int j = tid;
    const float pos = (float)(SD - 1 + (int)(r % SD));
    const float inv = __expf(-((float)(2 * j) / 64.f) * logf(10000.f));
    const float ang = pos * inv;
    const float sn = sinf(ang), cs = cosf(ang);
    const float x1 = xr[512 + 2 * j], x2 = xr[512 + 2 * j + 1];
    kpe[r * ROPEd + 2 * j]     = x1 * cs - x2 * sn;
    kpe[r * ROPEd + 2 * j + 1] = x1 * sn + x2 * cs;
  }
  __shared__ float red[256];
  float s = 0.f;
  for (int i = tid; i < CCd; i += 256) s += xr[i];
  red[tid] = s; __syncthreads();
  for (int off = 128; off > 0; off >>= 1) { if (tid < off) red[tid] += red[tid + off]; __syncthreads(); }
  const float mean = red[0] / (float)CCd; __syncthreads();
  float v = 0.f;
  for (int i = tid; i < CCd; i += 256) { float d = xr[i] - mean; v += d * d; }
  red[tid] = v; __syncthreads();
  for (int off = 128; off > 0; off >>= 1) { if (tid < off) red[tid] += red[tid + off]; __syncthreads(); }
  const float istd = rsqrtf(red[0] / (float)CCd + 1e-5f);
  for (int i = tid; i < CCd; i += 256)
    kvn[r * CCd + i] = (xr[i] - mean) * istd * g[i] + bb[i];
}

// ---------------- sinusoidal PE table Cpe[t][512] ----------------
__global__ __launch_bounds__(256) void cpe_kernel(float* __restrict__ Cpe) {
  const int p = blockIdx.x;
  for (int c = threadIdx.x; c < CCd; c += 256) {
    const int j2 = (c >> 1) * 2;
    const float dv = __expf((float)j2 * (-logf(10000.f) / (float)CCd));
    const float ang = (float)p * dv;
    Cpe[(long)p * CCd + c] = (c & 1) ? cosf(ang) : sinf(ang);
  }
}

// ---------------- prev_kv[b,t] = sig(C2[t]·P2[b,2t])*kvn[2t] + sig(C2[t]·P2[b,2t+1])*kvn[2t+1]
__global__ __launch_bounds__(128) void prevkv_kernel(
    const float* __restrict__ C2, const float* __restrict__ P2,
    const float* __restrict__ kvn, float* __restrict__ pkv)
{
  const int bt = blockIdx.x;
  const int b = bt >> 10, t = bt & 1023;
  const int tid = threadIdx.x;
  __shared__ float r0[128], r1[128];
  const float c2 = C2[(long)t * 128 + tid];
  const long row0 = (long)b * SD + 2 * t;
  r0[tid] = c2 * P2[row0 * 128 + tid];
  r1[tid] = c2 * P2[(row0 + 1) * 128 + tid];
  __syncthreads();
  for (int off = 64; off > 0; off >>= 1) {
    if (tid < off) { r0[tid] += r0[tid + off]; r1[tid] += r1[tid + off]; }
    __syncthreads();
  }
  const float w0 = 1.f / (1.f + __expf(-r0[0]));
  const float w1 = 1.f / (1.f + __expf(-r1[0]));
  const float* k0 = kvn + row0 * CCd;
  const float* k1 = kvn + (row0 + 1) * CCd;
  float* o = pkv + ((long)b * TT + t) * CCd;
  for (int c = tid; c < CCd; c += 128) o[c] = w0 * k0[c] + w1 * k1[c];
}

// ---------------- normalized kpe (t rows) broadcast into Kcat[...,128:192] (bf16) -------
__global__ __launch_bounds__(64) void kpen_kernel(
    const float* __restrict__ kpe, unsigned short* __restrict__ Kcat)
{
  const int bt = blockIdx.x;
  const int b = bt >> 10, t = bt & 1023;
  const int j = threadIdx.x;
  __shared__ float red[64];
  const float v = kpe[((long)b * SD + 2 * t + 1) * ROPEd + j];
  red[j] = v * v; __syncthreads();
  for (int off = 32; off > 0; off >>= 1) { if (j < off) red[j] += red[j + off]; __syncthreads(); }
  const unsigned short vn = f2bf(v / fmaxf(sqrtf(red[0]), 1e-12f));
  for (int h = 0; h < HD; ++h)
    Kcat[(((long)(b * HD + h)) * TT + t) * DKC + NOPEd + j] = vn;
}

// ---------------- Q prep: split heads, rope+normalize pe, fold pos_scale, emit bf16 ----
__global__ __launch_bounds__(256) void qprep_kernel(
    const float* __restrict__ q, const float* __restrict__ pos_scale,
    unsigned short* __restrict__ Qcat)
{
  const long r = blockIdx.x;                 // b*S + s
  const int b = (int)(r / SD), s = (int)(r % SD);
  const int tid = threadIdx.x;
  __shared__ float pe[HD][ROPEd];
  __shared__ float nrm2[HD];
  const float* qr = q + r * (HD * DKC);
  const float pos = (float)(SD - 1 + s);
  for (int idx = tid; idx < HD * 32; idx += 256) {
    const int h = idx >> 5, j = idx & 31;
    const float inv = __expf(-((float)(2 * j) / 64.f) * logf(10000.f));
    const float ang = pos * inv;
    const float sn = sinf(ang), cs = cosf(ang);
    const float x1 = qr[h * DKC + NOPEd + 2 * j];
    const float x2 = qr[h * DKC + NOPEd + 2 * j + 1];
    pe[h][2 * j]     = x1 * cs - x2 * sn;
    pe[h][2 * j + 1] = x1 * sn + x2 * cs;
  }
  __syncthreads();
  if (tid < HD) {                            // deterministic per-head norm
    float s2 = 0.f;
    for (int j = 0; j < ROPEd; ++j) s2 += pe[tid][j] * pe[tid][j];
    nrm2[tid] = s2;
  }
  __syncthreads();
  const float ps = *pos_scale;
  for (int idx = tid; idx < HD * DKC; idx += 256) {
    const int h = idx / DKC, d = idx % DKC;
    float v;
    if (d < NOPEd) v = qr[h * DKC + d];
    else           v = pe[h][d - NOPEd] * ps / fmaxf(sqrtf(nrm2[h]), 1e-12f);
    Qcat[(((long)(b * HD + h) * SD) + s) * DKC + d] = f2bf(v);
  }
}

// ---------------- fused flash attention per (b,h, 32-row block), bf16 inputs ----------
union AttnU {
  unsigned short Ks[64][192];   // QK stage
  unsigned short Ps[32][64];    // PV stage (Ks dead by then)
};

__global__ __launch_bounds__(256) void attn_kernel(
    const unsigned short* __restrict__ Qcat, const unsigned short* __restrict__ Kcat,
    const unsigned short* __restrict__ Vbuf, float* __restrict__ xo)
{
  const int bh = blockIdx.z;
  const int b = bh >> 4, h = bh & 15;
  const int s0 = blockIdx.x * 32;
  const int tid = threadIdx.x;
  const int wave = tid >> 5, lane = tid & 31;
  const int wM = wave & 1;      // 2 waves along M (16 rows)
  const int wN = wave >> 1;     // 4 waves along N

  __shared__ __align__(16) unsigned short Qs[32][192];
  __shared__ __align__(16) AttnU u;
  __shared__ __align__(16) unsigned short Vts[128][64];   // V already (d,t) in global
  __shared__ float Ss[32][64];
  __shared__ float mrow[32], lrow[32], arow[32];

  // async copy of Q tile (32x192 bf16, contiguous rows == LDS layout)
  {
    const unsigned short* qg = Qcat + ((long)bh * SD + s0) * DKC;
    const unsigned q0 = ldsoff(&Qs[0][0]);
    for (int c = tid; c < (32 * DKC * 2) / 16; c += 256)   // 768 x 16B chunks
      async_b128(q0 + c * 16, (unsigned)(c * 16), qg);
  }
  if (tid < 32) { mrow[tid] = -1e30f; lrow[tid] = 0.f; }
  wait_async0();
  __syncthreads();

  floatx8 oacc[2] = {};
  const float scale = 0.0721687836f;   // 192^-0.5

  for (int t0 = 0; t0 < TT; t0 += 64) {
    // async copy K tile (64x192, contiguous) and V tile (128 rows x 64, strided)
    {
      const unsigned short* kg = Kcat + ((long)bh * TT + t0) * DKC;
      const unsigned k0 = ldsoff(&u.Ks[0][0]);
      for (int c = tid; c < (64 * DKC * 2) / 16; c += 256)  // 1536 chunks
        async_b128(k0 + c * 16, (unsigned)(c * 16), kg);
      const unsigned short* vg = Vbuf + ((long)bh * VHd) * TT + t0;
      const unsigned v0 = ldsoff(&Vts[0][0]);
      for (int c = tid; c < (128 * 64 * 2) / 16; c += 256) { // 1024 chunks
        const int row = c >> 3, kk = c & 7;                  // row=d, 8 chunks/row
        async_b128(v0 + c * 16, (unsigned)(row * (TT * 2) + kk * 16), vg);
      }
    }
    wait_async0();
    __syncthreads();

    // S = Q · K^T  (each wave one 16x16 tile of the 32x64 score block)
    floatx8 sacc = {};
    #pragma unroll
    for (int ks = 0; ks < 6; ++ks) {
      bf16x16 aq = load_frag(&Qs[wM * 16][ks * 32], 192);
      bf16x16 bk = load_frag(&u.Ks[wN * 16][ks * 32], 192);
      sacc = wmma_bf16(aq, bk, sacc);
    }
    {
      const int cN = lane & 15, rB = (lane >> 4) << 3;
      #pragma unroll
      for (int e = 0; e < 8; ++e)
        Ss[wM * 16 + rB + e][wN * 16 + cN] = sacc[e] * scale;
    }
    __syncthreads();

    // online softmax update, one thread per row
    if (tid < 32) {
      const float mo = mrow[tid];
      float cm = -1e30f;
      for (int j = 0; j < 64; ++j) cm = fmaxf(cm, Ss[tid][j]);
      const float mn = fmaxf(mo, cm);
      const float al = __expf(mo - mn);
      float sum = 0.f;
      for (int j = 0; j < 64; ++j) { float p = __expf(Ss[tid][j] - mn); sum += p; Ss[tid][j] = p; }
      mrow[tid] = mn;
      lrow[tid] = lrow[tid] * al + sum;
      arow[tid] = al;
    }
    __syncthreads();

    // P (bf16, reuses Ks LDS) ; rescale running accumulators
    for (int idx = tid; idx < 32 * 64; idx += 256) {
      const int r = idx >> 6, c = idx & 63;
      u.Ps[r][c] = f2bf(Ss[r][c]);
    }
    {
      const int rB = (lane >> 4) << 3;
      #pragma unroll
      for (int e = 0; e < 8; ++e) {
        const float al = arow[wM * 16 + rB + e];
        oacc[0][e] *= al; oacc[1][e] *= al;
      }
    }
    __syncthreads();

    // O += P · V  (wave covers 16M x 32N)
    #pragma unroll
    for (int kt = 0; kt < 2; ++kt) {
      bf16x16 ap = load_frag(&u.Ps[wM * 16][kt * 32], 64);
      #pragma unroll
      for (int j = 0; j < 2; ++j) {
        bf16x16 bv = load_frag(&Vts[wN * 32 + j * 16][kt * 32], 64);
        oacc[j] = wmma_bf16(ap, bv, oacc[j]);
      }
    }
    __syncthreads();
  }

  {
    const int cN = lane & 15, rB = (lane >> 4) << 3;
    #pragma unroll
    for (int j = 0; j < 2; ++j)
      #pragma unroll
      for (int e = 0; e < 8; ++e) {
        const int r = wM * 16 + rB + e;
        const int c = wN * 32 + j * 16 + cN;
        xo[((long)b * SD + s0 + r) * (HD * VHd) + h * VHd + c] = oacc[j][e] / lrow[r];
      }
  }
}

// ---------------- host orchestration ----------------
extern "C" void kernel_launch(void* const* d_in, const int* in_sizes, int n_in,
                              void* d_out, int out_size, void* d_ws, size_t ws_size,
                              hipStream_t stream) {
  (void)in_sizes; (void)n_in; (void)out_size; (void)ws_size;
  const float* query   = (const float*)d_in[0];
  const float* key     = (const float*)d_in[1];
  /* value d_in[2] unused by reference */
  const float* wq_a    = (const float*)d_in[3];
  const float* q_ln_g  = (const float*)d_in[4];
  const float* q_ln_b  = (const float*)d_in[5];
  const float* wq_b    = (const float*)d_in[6];
  const float* wkv_a   = (const float*)d_in[7];
  const float* kv_ln_g = (const float*)d_in[8];
  const float* kv_ln_b = (const float*)d_in[9];
  const float* wkv_b   = (const float*)d_in[10];
  const float* wo      = (const float*)d_in[11];
  const float* fc_c_w  = (const float*)d_in[12];
  const float* fc_c_b  = (const float*)d_in[13];
  const float* fc_p_w  = (const float*)d_in[14];
  const float* fc_p_b  = (const float*)d_in[15];
  const float* pos_scale = (const float*)d_in[16];
  float* out = (float*)d_out;

  char* wp = (char*)d_ws;
  auto alloc = [&](size_t bytes) -> void* {
    void* p = (void*)wp;
    wp += (bytes + 255) & ~size_t(255);
    return p;
  };
  const long BS = (long)BD * SD;                               // 4096 rows
  float* ql   = (float*)alloc(BS * QRd * 4);
  float* qbuf = (float*)alloc(BS * (HD * DKC) * 4);            // reused as xo later
  float* kvf  = (float*)alloc(BS * 576 * 4);
  float* kvn  = (float*)alloc(BS * CCd * 4);
  float* kpe  = (float*)alloc(BS * ROPEd * 4);
  float* P2   = (float*)alloc(BS * 128 * 4);
  float* Cpe  = (float*)alloc((long)TT * CCd * 4);
  float* C2   = (float*)alloc((long)TT * 128 * 4);
  float* pkv  = (float*)alloc((long)BD * TT * CCd * 4);
  unsigned short* Kcat = (unsigned short*)alloc((long)BD * HD * TT * DKC * 2);
  unsigned short* Vb   = (unsigned short*)alloc((long)BD * HD * VHd * TT * 2);
  unsigned short* Qcat = (unsigned short*)alloc((long)BD * HD * SD * DKC * 2);
  float* xo   = qbuf;                                          // qbuf dead after qprep

  const dim3 blk(256);
  // 1. ql = query @ wq_a^T
  gemm_wmma_kernel<0><<<dim3(QRd/128, BS/128, 1), blk, 0, stream>>>(
      query, wq_a, nullptr, ql, (int)BS, QRd, EDIM, 0, 0, 0, nullptr, nullptr);
  // 2. LN(ql) in place
  ln_kernel<<<dim3((unsigned)BS), blk, 0, stream>>>(ql, q_ln_g, q_ln_b, ql, QRd);
  // 3. q = ql @ wq_b^T
  gemm_wmma_kernel<0><<<dim3((HD*DKC)/128, BS/128, 1), blk, 0, stream>>>(
      ql, wq_b, nullptr, qbuf, (int)BS, HD*DKC, QRd, 0, 0, 0, nullptr, nullptr);
  // 4. kvf = key @ wkv_a^T  (N=576 ragged)
  gemm_wmma_kernel<0><<<dim3((576+127)/128, BS/128, 1), blk, 0, stream>>>(
      key, wkv_a, nullptr, kvf, (int)BS, 576, EDIM, 0, 0, 0, nullptr, nullptr);
  // 5. kvn = LN(kvf[:, :512]); kpe = rope(kvf[:, 512:])
  kvprep_kernel<<<dim3((unsigned)BS), blk, 0, stream>>>(kvf, kv_ln_g, kv_ln_b, kvn, kpe);
  // 6. P2 = kvn @ fc_p_w^T + b
  gemm_wmma_kernel<1><<<dim3(1, BS/128, 1), blk, 0, stream>>>(
      kvn, fc_p_w, fc_p_b, P2, (int)BS, 128, CCd, 0, 0, 0, nullptr, nullptr);
  // 7. Cpe table; 8. C2 = Cpe @ fc_c_w^T + b
  cpe_kernel<<<dim3(TT), blk, 0, stream>>>(Cpe);
  gemm_wmma_kernel<1><<<dim3(1, TT/128, 1), blk, 0, stream>>>(
      Cpe, fc_c_w, fc_c_b, C2, TT, 128, CCd, 0, 0, 0, nullptr, nullptr);
  // 9. prev_kv combine
  prevkv_kernel<<<dim3(BD*TT), dim3(128), 0, stream>>>(C2, P2, kvn, pkv);
  // 10. ck/cv absorption: per (b,h): prev_kv @ wkvb[h]^T -> Kcat[:,:128] (bf16), V^T (bf16)
  gemm_wmma_kernel<2><<<dim3(2, TT/128, BD*HD), blk, 0, stream>>>(
      pkv, wkv_b, nullptr, nullptr, TT, 256, CCd,
      (long)TT*CCd, (long)256*CCd, 0, Kcat, Vb);
  // 11. normalized kpe -> Kcat[..., 128:192]
  kpen_kernel<<<dim3(BD*TT), dim3(64), 0, stream>>>(kpe, Kcat);
  // 12. Qcat = [q_nope, pos_scale * normalize(rope(q_pe))] (bf16)
  qprep_kernel<<<dim3((unsigned)BS), blk, 0, stream>>>(qbuf, pos_scale, Qcat);
  // 13. fused flash attention (async LDS copies) -> xo (B,S,H*128) f32
  attn_kernel<<<dim3(SD/32, 1, BD*HD), blk, 0, stream>>>(Qcat, Kcat, Vb, xo);
  // 14. out = xo @ wo^T
  gemm_wmma_kernel<0><<<dim3(EDIM/128, BS/128, 1), blk, 0, stream>>>(
      xo, wo, nullptr, out, (int)BS, EDIM, HD*VHd, 0, 0, 0, nullptr, nullptr);
}
